// CARAFEPlusPlusUpsample_78348793414242
// MI455X (gfx1250) — matrix operand order repaired and checked by reading, not compiled, not run on hardware
//
#include <hip/hip_runtime.h>
#include <hip/hip_bf16.h>

// ---------------------------------------------------------------------------
// CARAFE++ upsample for MI455X (gfx1250, wave32, WMMA + TDM).
//   x:      (2, 256, 40, 40)  f32
//   w_comp: (64, 256, 1, 1)   f32
//   w_enc:  (100, 64, 3, 3)   f32
//   out:    (2, 256, 80, 80)  f32
// Pipeline: prep(bf16 convert/transpose) -> WMMA GEMM1 (compress, B tile
// staged to LDS via tensor_load_to_lds) -> im2col -> WMMA GEMM2 (encoder
// conv, same TDM staging) -> softmax/pixelshuffle -> reassembly.
// ---------------------------------------------------------------------------

typedef __attribute__((ext_vector_type(8)))  __bf16 v8bf;
typedef __attribute__((ext_vector_type(16))) __bf16 v16bf;
typedef __attribute__((ext_vector_type(8)))  float  v8f;
typedef __attribute__((ext_vector_type(4))) unsigned int ui32x4;
typedef __attribute__((ext_vector_type(8))) int i32x8;
typedef __attribute__((ext_vector_type(4))) int i32x4;

#define NB   2
#define C_IN 256
#define C_MD 64
#define HH   40
#define WW   40
#define PP   (HH*WW)           // 1600
#define K2   25
#define ENCC 100
#define MPAD 112               // 100 padded to 7*16
#define KCOL 576               // 9*64 im2col K
#define HU   80
#define WU   80
#define QQ   (HU*WU)           // 6400

__device__ __forceinline__ unsigned short f2bf(float f) {
    unsigned int u = __float_as_uint(f);
    unsigned int r = u + 0x7FFFu + ((u >> 16) & 1u);   // round-to-nearest-even
    return (unsigned short)(r >> 16);
}

// Build a 16-element bf16 fragment from two contiguous 16-byte chunks.
__device__ __forceinline__ v16bf load_frag(const unsigned short* base, int hi_off) {
    v8bf lo = *(const v8bf*)(base);
    v8bf hi = *(const v8bf*)(base + hi_off);
    v16bf a;
#pragma unroll
    for (int i = 0; i < 8; ++i) { a[i] = lo[i]; a[i + 8] = hi[i]; }
    return a;
}

// ---------------------------------------------------------------------------
// TDM: 1D contiguous copy of `nelem` bf16 elements global -> LDS.
// D# layout per CDNA5 ISA ch.10.8 (group0 128b, group1 256b, groups2/3 zero).
//   group0: [1:0]=count=1, [63:32]=lds_addr, [120:64]=global_addr, [127:126]=2
//   group1: [17:16]=data_size(1=2B), [79:48]=tensor_dim0, [111:80]=tensor_dim1,
//           [127:112]=tile_dim0, [143:128]=tile_dim1, [207:160]=dim0_stride
// ---------------------------------------------------------------------------
__device__ __forceinline__ void tdm_load_1d(unsigned int lds_addr,
                                            unsigned long long gaddr,
                                            unsigned int nelem) {
    ui32x4 g0;
    g0.x = 1u;                                                // count=1, user
    g0.y = lds_addr;                                          // lds_addr
    g0.z = (unsigned int)(gaddr & 0xFFFFFFFFu);               // gaddr[31:0]
    g0.w = (unsigned int)((gaddr >> 32) & 0x1FFFFFFu) | (2u << 30); // type=2

    i32x8 g1;
    g1[0] = (int)(1u << 16);                                  // data_size = 2B
    g1[1] = (int)((nelem & 0xFFFFu) << 16);                   // tensor_dim0 lo16
    g1[2] = (int)(((nelem >> 16) & 0xFFFFu) | (1u << 16));    // dim0 hi16 | tensor_dim1=1
    g1[3] = (int)((nelem & 0xFFFFu) << 16);                   // tile_dim0 (<=64K)
    g1[4] = 1;                                                // tile_dim1 = 1
    g1[5] = (int)nelem;                                       // tensor_dim0_stride lo32
    g1[6] = 0;                                                // stride hi | dim1_stride lo
    g1[7] = 0;

    i32x4 gz; gz[0] = 0; gz[1] = 0; gz[2] = 0; gz[3] = 0;
#if __clang_major__ >= 23
    i32x8 gp; gp[0]=0; gp[1]=0; gp[2]=0; gp[3]=0; gp[4]=0; gp[5]=0; gp[6]=0; gp[7]=0;
    __builtin_amdgcn_tensor_load_to_lds(g0, g1, gz, gz, gp, 0);
#else
    __builtin_amdgcn_tensor_load_to_lds(g0, g1, gz, gz, 0);
#endif
}

// ---------------------------------------------------------------------------
// Kernel 0: convert x (NCHW f32) -> xt (b,p,c) bf16; w_comp -> bf16 (64x256);
//           w_enc -> bf16 reordered (112 x 576), K layout = offset*64 + c.
// ---------------------------------------------------------------------------
#define N_XT   (NB*PP*C_IN)          // 819200
#define N_WC   (C_MD*C_IN)           // 16384
#define N_WE   (MPAD*KCOL)           // 64512
#define N_PREP (N_XT + N_WC + N_WE)

__global__ void prep_convert(const float* __restrict__ x,
                             const float* __restrict__ w_comp,
                             const float* __restrict__ w_enc,
                             unsigned short* __restrict__ xt,
                             unsigned short* __restrict__ wcb,
                             unsigned short* __restrict__ wer) {
    for (int i = blockIdx.x * blockDim.x + threadIdx.x; i < N_PREP;
         i += gridDim.x * blockDim.x) {
        if (i < N_XT) {
            int b = i / (PP * C_IN);
            int rem = i % (PP * C_IN);
            int p = rem / C_IN;
            int c = rem % C_IN;
            xt[i] = f2bf(x[((size_t)b * C_IN + c) * PP + p]);
        } else if (i < N_XT + N_WC) {
            int j = i - N_XT;
            wcb[j] = f2bf(w_comp[j]);
        } else {
            int j = i - N_XT - N_WC;
            int m = j / KCOL;
            int k = j % KCOL;
            int o = k >> 6;        // 0..8 kernel offset
            int c = k & 63;        // 0..63 channel
            wer[j] = (m < ENCC) ? f2bf(w_enc[(size_t)m * 576 + c * 9 + o])
                                : (unsigned short)0;
        }
    }
}

// ---------------------------------------------------------------------------
// Kernel 1: compress GEMM.  ct[b][p][m] = sum_c wcb[m][c] * xt[b][p][c]
// M=64 (4 tiles, one per wave), N=1600 (16/block), K=256 (8 steps of 32).
// B tile (16 x 256 bf16 = 8 KB, shared by all 4 waves) staged via TDM.
// ---------------------------------------------------------------------------
__global__ __launch_bounds__(128)
void gemm_compress(const unsigned short* __restrict__ wcb,   // 64 x 256
                   const unsigned short* __restrict__ xt,    // B x 1600 x 256
                   unsigned short* __restrict__ ct) {        // B x 1600 x 64
    __shared__ __align__(16) unsigned short ldsB[16 * C_IN];

    const int ntile = blockIdx.x;          // 0..99
    const int b     = blockIdx.z;
    const int wave  = threadIdx.x >> 5;    // M tile 0..3
    const int lane  = threadIdx.x & 31;
    const int half  = lane >> 4;
    const int lm    = lane & 15;

    if (wave == 0) {
        unsigned long long g = (unsigned long long)(uintptr_t)(
            xt + ((size_t)b * PP + ntile * 16) * C_IN);
        tdm_load_1d((unsigned int)(uintptr_t)ldsB, g, 16 * C_IN);
        __builtin_amdgcn_s_wait_tensorcnt(0);
    }
    __syncthreads();

    const unsigned short* arow = wcb + (size_t)(wave * 16 + lm) * C_IN;
    const unsigned short* brow = ldsB + lm * C_IN;

    v8f acc = {};
#pragma unroll
    for (int kk = 0; kk < C_IN; kk += 32) {
        // A 16x32 bf16: lane-half selects K base 8*half; chunks at +0 and +16.
        v16bf a = load_frag(arow + kk + 8 * half, 16);
        // B 32x16 bf16: lane-half selects K base 8*half; chunks at +0 and +16.
        v16bf bm = load_frag(brow + kk + 8 * half, 16);
        acc = __builtin_amdgcn_wmma_f32_16x16x32_bf16(
            false, a, false, bm, (short)0, acc, false, false);
    }

    // D tile: lane(half,n=lm) holds M = wave*16 + 8*half + r, r=0..7 (contiguous).
    union { unsigned short us[8]; uint4 v; } u;
#pragma unroll
    for (int r = 0; r < 8; ++r) u.us[r] = f2bf(acc[r]);
    unsigned short* dst =
        ct + ((size_t)b * PP + ntile * 16 + lm) * C_MD + wave * 16 + 8 * half;
    *(uint4*)dst = u.v;
}

// ---------------------------------------------------------------------------
// Kernel 2: im2col for the 3x3 encoder conv (pad 1), K layout offset*64+c.
// Each unit copies 4 bf16 (uint2).
// ---------------------------------------------------------------------------
#define N_I2C (NB*PP*9*16)    // 460800 units
__global__ void im2col(const unsigned short* __restrict__ ct,   // B x 1600 x 64
                       unsigned short* __restrict__ col) {      // B x 1600 x 576
    for (int i = blockIdx.x * blockDim.x + threadIdx.x; i < N_I2C;
         i += gridDim.x * blockDim.x) {
        int cu = i & 15;            // 4-element chunk -> c = cu*4
        int o  = (i >> 4) % 9;
        int rest = i / 144;
        int p = rest % PP;
        int b = rest / PP;
        int ki = o / 3, kj = o % 3;
        int ph = p / WW, pw = p % WW;
        int rr = ph + ki - 1, cs = pw + kj - 1;
        unsigned short* dst =
            col + ((size_t)b * PP + p) * KCOL + o * 64 + cu * 4;
        if (rr >= 0 && rr < HH && cs >= 0 && cs < WW) {
            const unsigned short* src =
                ct + ((size_t)b * PP + rr * WW + cs) * C_MD + cu * 4;
            *(uint2*)dst = *(const uint2*)src;
        } else {
            uint2 z; z.x = 0u; z.y = 0u;
            *(uint2*)dst = z;
        }
    }
}

// ---------------------------------------------------------------------------
// Kernel 3: encoder GEMM.  enc[b][m][p] = sum_k wer[m][k] * col[b][p][k]
// M=112 (7 tiles, one per wave), N=1600 (16/block), K=576 (18 steps of 32).
// B tile (16 x 576 bf16 = 18 KB, shared by all 7 waves) staged via TDM.
// ---------------------------------------------------------------------------
__global__ __launch_bounds__(224)
void gemm_encode(const unsigned short* __restrict__ wer,    // 112 x 576
                 const unsigned short* __restrict__ col,    // B x 1600 x 576
                 float* __restrict__ enc) {                 // B x 100 x 1600
    __shared__ __align__(16) unsigned short ldsB[16 * KCOL];

    const int ntile = blockIdx.x;
    const int b     = blockIdx.z;
    const int wave  = threadIdx.x >> 5;    // M tile 0..6
    const int lane  = threadIdx.x & 31;
    const int half  = lane >> 4;
    const int lm    = lane & 15;

    if (wave == 0) {
        unsigned long long g = (unsigned long long)(uintptr_t)(
            col + ((size_t)b * PP + ntile * 16) * KCOL);
        tdm_load_1d((unsigned int)(uintptr_t)ldsB, g, 16 * KCOL);
        __builtin_amdgcn_s_wait_tensorcnt(0);
    }
    __syncthreads();

    const unsigned short* arow = wer + (size_t)(wave * 16 + lm) * KCOL;
    const unsigned short* brow = ldsB + lm * KCOL;

    v8f acc = {};
#pragma unroll 3
    for (int kk = 0; kk < KCOL; kk += 32) {
        v16bf a  = load_frag(arow + kk + 8 * half, 16);
        v16bf bm = load_frag(brow + kk + 8 * half, 16);
        acc = __builtin_amdgcn_wmma_f32_16x16x32_bf16(
            false, a, false, bm, (short)0, acc, false, false);
    }

    const int n = ntile * 16 + lm;
#pragma unroll
    for (int r = 0; r < 8; ++r) {
        int m = wave * 16 + 8 * half + r;
        if (m < ENCC) enc[((size_t)b * ENCC + m) * PP + n] = acc[r];
    }
}

// ---------------------------------------------------------------------------
// Kernel 4: pixel-shuffle + softmax over the 25 reassembly weights.
// kern[b][k][q], q = r*80+cc, source channel k*4 + (r&1)*2 + (cc&1).
// ---------------------------------------------------------------------------
__global__ void softmax_shuffle(const float* __restrict__ enc,
                                float* __restrict__ kern) {
    int idx = blockIdx.x * blockDim.x + threadIdx.x;   // b*QQ
    if (idx >= NB * QQ) return;
    int b = idx / QQ, q = idx % QQ;
    int r = q / WU, cc = q % WU;
    int sub = (r & 1) * 2 + (cc & 1);
    int p = (r >> 1) * WW + (cc >> 1);

    float v[K2];
    float mx = -3.4e38f;
#pragma unroll
    for (int k = 0; k < K2; ++k) {
        v[k] = enc[((size_t)b * ENCC + k * 4 + sub) * PP + p];
        mx = fmaxf(mx, v[k]);
    }
    float s = 0.f;
#pragma unroll
    for (int k = 0; k < K2; ++k) { v[k] = __expf(v[k] - mx); s += v[k]; }
    float inv = 1.f / s;
#pragma unroll
    for (int k = 0; k < K2; ++k)
        kern[((size_t)b * K2 + k) * QQ + q] = v[k] * inv;
}

// ---------------------------------------------------------------------------
// Kernel 5: reassembly.  out[b][c][q] = sum_t x[b][c][nbr_t(p)] * kern[b][t][q]
// with p = q >> 2 (faithful to the torch "efficient" flattening).
// OOB taps get weight 0 and a clamped (in-bounds) offset -> branch-free FMAs.
// ---------------------------------------------------------------------------
__global__ __launch_bounds__(256)
void reassemble(const float* __restrict__ x,
                const float* __restrict__ kern,
                float* __restrict__ out) {
    int idx = blockIdx.x * blockDim.x + threadIdx.x;   // b*QQ
    if (idx >= NB * QQ) return;
    int b = idx / QQ, q = idx % QQ;
    int p = q >> 2;
    int ph = p / WW, pw = p % WW;

    float w[K2];
    int   off[K2];
#pragma unroll
    for (int ki = 0; ki < 5; ++ki) {
#pragma unroll
        for (int kj = 0; kj < 5; ++kj) {
            int t = ki * 5 + kj;
            int rr = ph + ki - 2, cs = pw + kj - 2;
            bool ok = (rr >= 0) && (rr < HH) && (cs >= 0) && (cs < WW);
            w[t]   = ok ? kern[((size_t)b * K2 + t) * QQ + q] : 0.f;
            off[t] = ok ? rr * WW + cs : 0;
        }
    }

    const float* xb = x + (size_t)b * C_IN * PP;
    float* ob = out + (size_t)b * C_IN * QQ + q;
#pragma unroll 4
    for (int c = 0; c < C_IN; ++c) {
        const float* xc = xb + (size_t)c * PP;
        float acc = 0.f;
#pragma unroll
        for (int t = 0; t < K2; ++t) acc = fmaf(xc[off[t]], w[t], acc);
        ob[(size_t)c * QQ] = acc;
    }
}

// ---------------------------------------------------------------------------
extern "C" void kernel_launch(void* const* d_in, const int* in_sizes, int n_in,
                              void* d_out, int out_size, void* d_ws, size_t ws_size,
                              hipStream_t stream) {
    const float* x      = (const float*)d_in[0];
    const float* w_comp = (const float*)d_in[1];
    const float* w_enc  = (const float*)d_in[2];
    float* out = (float*)d_out;

    // Workspace carve-up (256B aligned).
    size_t o = 0;
    auto carve = [&](size_t bytes) { size_t r = o; o += (bytes + 255) & ~(size_t)255; return r; };
    size_t off_xt   = carve((size_t)N_XT * 2);                 // bf16 x, (b,p,c)
    size_t off_wcb  = carve((size_t)N_WC * 2);                 // bf16 w_comp
    size_t off_wer  = carve((size_t)N_WE * 2);                 // bf16 w_enc reordered
    size_t off_ct   = carve((size_t)NB * PP * C_MD * 2);       // bf16 compressed
    size_t off_col  = carve((size_t)NB * PP * KCOL * 2);       // bf16 im2col
    size_t off_enc  = carve((size_t)NB * ENCC * PP * 4);       // f32 encoder out
    size_t off_kern = carve((size_t)NB * K2 * QQ * 4);         // f32 softmaxed kernels
    if (ws_size < o) return;   // insufficient workspace (deterministic no-op)

    char* ws = (char*)d_ws;
    unsigned short* xt  = (unsigned short*)(ws + off_xt);
    unsigned short* wcb = (unsigned short*)(ws + off_wcb);
    unsigned short* wer = (unsigned short*)(ws + off_wer);
    unsigned short* ct  = (unsigned short*)(ws + off_ct);
    unsigned short* col = (unsigned short*)(ws + off_col);
    float* enc  = (float*)(ws + off_enc);
    float* kern = (float*)(ws + off_kern);

    // 0) precision convert + layout transforms
    prep_convert<<<(N_PREP + 255) / 256, 256, 0, stream>>>(x, w_comp, w_enc,
                                                           xt, wcb, wer);
    // 1) channel compression GEMM (WMMA bf16, TDM-staged B tile)
    gemm_compress<<<dim3(PP / 16, 1, NB), 128, 0, stream>>>(wcb, xt, ct);
    // 2) im2col for 3x3 conv
    im2col<<<(N_I2C + 255) / 256, 256, 0, stream>>>(ct, col);
    // 3) encoder GEMM (WMMA bf16, TDM-staged B tile)
    gemm_encode<<<dim3(PP / 16, 1, NB), 224, 0, stream>>>(wer, col, enc);
    // 4) pixel shuffle + softmax
    softmax_shuffle<<<(NB * QQ + 255) / 256, 256, 0, stream>>>(enc, kern);
    // 5) reassembly -> output
    reassemble<<<(NB * QQ + 255) / 256, 256, 0, stream>>>(x, kern, out);
}
